// CrossAttention_7627861918199
// MI455X (gfx1250) — compile-verified, matrix operand order checked by reading, not run on hardware
//
#include <hip/hip_runtime.h>

// ---------------------------------------------------------------------------
// CrossAttention for MI455X (gfx1250, wave32, WMMA bf16 w/ f32 accumulate)
// ---------------------------------------------------------------------------
//   1) fp32 -> bf16 conversion of x, context, Wq, Wkv, Wo
//   2) Q  = xb @ Wq^T      (WMMA gemm, 32x32 tile/wave, bf16 out)
//   3) KV = cb @ Wkv^T     (WMMA gemm; K row-major, V stored transposed
//                           vt[b, h*64+d, NKV])
//   4) flash attention: block = 4 waves sharing (b,h) and a 64-row q block;
//      K/V tiles staged to LDS per 32-KV step (async-to-LDS when available),
//      S=QK^T via WMMA, online softmax, P @ V via WMMA
//   5) out = O @ Wo^T + bo (WMMA gemm, fp32 out)
// ---------------------------------------------------------------------------

typedef __attribute__((ext_vector_type(16))) __bf16 v16bf;
typedef __attribute__((ext_vector_type(8)))  float  v8f;

#define N_HEADS 8
#define D_HEAD  64
#define QDIM    512
#define INNER   512      // N_HEADS * D_HEAD
#define BATCH   4
#define NQ      1024
#define NKV     4096

static __device__ __forceinline__ v8f wmma_bf16(v16bf a, v16bf b, v8f c) {
  // (neg_a, A, neg_b, B, c_mod, C, reuse_a, reuse_b)
  return __builtin_amdgcn_wmma_f32_16x16x32_bf16(false, a, false, b,
                                                 (short)0, c, false, false);
}

// 16x32 bf16 fragment (A layout; also B^T given 16 rows x K row-major).
// ISA 7.12.2: lane L -> row L&15; elems 0..7 = K (L>>4)*8+0..7,
// elems 8..15 = K 16+(L>>4)*8+0..7.  Two b128 loads per lane.
static __device__ __forceinline__ v16bf load_frag(const __bf16* p, int ld, int lane) {
  const int row  = lane & 15;
  const int half = lane >> 4;
  const __bf16* r = p + (size_t)row * ld + half * 8;
  union { v16bf v; uint4 u[2]; } t;
  t.u[0] = *(const uint4*)(r);
  t.u[1] = *(const uint4*)(r + 16);
  return t.v;
}

// C[32,32] = A[32,K] * B[32,K]^T, fp32 accum.  acc[mi*2+ni].
static __device__ __forceinline__ void gemm_tile32(const __bf16* A, const __bf16* B,
                                                   int lda, int ldb, int K, int lane,
                                                   v8f acc[4]) {
#pragma unroll
  for (int t = 0; t < 4; ++t)
#pragma unroll
    for (int i = 0; i < 8; ++i) acc[t][i] = 0.0f;
  for (int k = 0; k < K; k += 32) {
    v16bf a0 = load_frag(A + k, lda, lane);
    v16bf a1 = load_frag(A + (size_t)16 * lda + k, lda, lane);
    v16bf b0 = load_frag(B + k, ldb, lane);
    v16bf b1 = load_frag(B + (size_t)16 * ldb + k, ldb, lane);
    acc[0] = wmma_bf16(a0, b0, acc[0]);
    acc[1] = wmma_bf16(a0, b1, acc[1]);
    acc[2] = wmma_bf16(a1, b0, acc[2]);
    acc[3] = wmma_bf16(a1, b1, acc[3]);
  }
}

// ---- async global->LDS staging (gfx1250), with portable fallback ----------
// Builtin signature (from hipcc diagnostics): native int vectors
// (vector_size(16)), global (as1) src first, LDS (as3) dst second.
typedef int ivec4 __attribute__((vector_size(16)));

static __device__ __forceinline__ void stage_b128(const __bf16* g, __bf16* l) {
#if __has_builtin(__builtin_amdgcn_global_load_async_to_lds_b128)
  __builtin_amdgcn_global_load_async_to_lds_b128(
      (__attribute__((address_space(1))) ivec4*)g,
      (__attribute__((address_space(3))) ivec4*)l, 0, 0);
#else
  *(uint4*)l = *(const uint4*)g;
#endif
}
static __device__ __forceinline__ void stage_wait() {
#if __has_builtin(__builtin_amdgcn_global_load_async_to_lds_b128)
#if __has_builtin(__builtin_amdgcn_s_wait_asynccnt)
  __builtin_amdgcn_s_wait_asynccnt(0);
#else
  asm volatile("s_wait_asynccnt 0" ::: "memory");
#endif
#endif
}

// ---------------------------------------------------------------------------
__global__ void cvt_f32_bf16x2(const float2* __restrict__ in,
                               unsigned int* __restrict__ out, int n2) {
  int i = blockIdx.x * blockDim.x + threadIdx.x;
  if (i < n2) {
    float2 v = in[i];
    union { __bf16 h[2]; unsigned int u; } t;
    t.h[0] = (__bf16)v.x;
    t.h[1] = (__bf16)v.y;
    out[i] = t.u;
  }
}

// Q = xb @ Wq^T : M=B*NQ, N=INNER, K=QDIM.  wave = 32x32 tile.
__global__ __launch_bounds__(128)
void proj_q_kernel(const __bf16* __restrict__ xb, const __bf16* __restrict__ wq,
                   __bf16* __restrict__ qb) {
  const int lane = threadIdx.x & 31, wave = threadIdx.x >> 5;
  const int m0 = blockIdx.x * 32;
  const int n0 = (blockIdx.y * 4 + wave) * 32;
  v8f acc[4];
  gemm_tile32(xb + (size_t)m0 * QDIM, wq + (size_t)n0 * QDIM, QDIM, QDIM, QDIM, lane, acc);
  const int half = lane >> 4, col = lane & 15;
#pragma unroll
  for (int mi = 0; mi < 2; ++mi)
#pragma unroll
    for (int ni = 0; ni < 2; ++ni)
#pragma unroll
      for (int r = 0; r < 8; ++r)
        qb[(size_t)(m0 + mi * 16 + half * 8 + r) * INNER + n0 + ni * 16 + col] =
            (__bf16)acc[mi * 2 + ni][r];
}

// KV = cb @ Wkv^T : M=B*NKV, N=2*INNER.  n<INNER -> K row-major;
// n>=INNER -> V stored transposed: vt[(b*INNER + (n-INNER))*NKV + j].
__global__ __launch_bounds__(128)
void proj_kv_kernel(const __bf16* __restrict__ cb, const __bf16* __restrict__ wkv,
                    __bf16* __restrict__ kb, __bf16* __restrict__ vt) {
  const int lane = threadIdx.x & 31, wave = threadIdx.x >> 5;
  const int m0 = blockIdx.x * 32;
  const int n0 = (blockIdx.y * 4 + wave) * 32;
  v8f acc[4];
  gemm_tile32(cb + (size_t)m0 * QDIM, wkv + (size_t)n0 * QDIM, QDIM, QDIM, QDIM, lane, acc);
  const int half = lane >> 4, col = lane & 15;
  if (n0 < INNER) {
#pragma unroll
    for (int mi = 0; mi < 2; ++mi)
#pragma unroll
      for (int ni = 0; ni < 2; ++ni)
#pragma unroll
        for (int r = 0; r < 8; ++r)
          kb[(size_t)(m0 + mi * 16 + half * 8 + r) * INNER + n0 + ni * 16 + col] =
              (__bf16)acc[mi * 2 + ni][r];
  } else {
#pragma unroll
    for (int mi = 0; mi < 2; ++mi)
#pragma unroll
      for (int ni = 0; ni < 2; ++ni) {
        const int nn = n0 - INNER + ni * 16 + col;   // h*64 + d
#pragma unroll
        for (int r = 0; r < 8; ++r) {
          const int row = m0 + mi * 16 + half * 8 + r;   // b*NKV + j
          const int b = row >> 12;
          const int j = row & (NKV - 1);
          vt[((size_t)b * INNER + nn) * NKV + j] = (__bf16)acc[mi * 2 + ni][r];
        }
      }
  }
}

// Flash attention: block = 128 threads (4 waves) sharing (b, h); wave w owns
// q rows [qblk*64 + w*16, +16).  K/V tiles staged through LDS per 32-KV step.
__global__ __launch_bounds__(128)
void attn_kernel(const __bf16* __restrict__ qb, const __bf16* __restrict__ kb,
                 const __bf16* __restrict__ vt, const unsigned char* __restrict__ mask,
                 __bf16* __restrict__ ob) {
  __shared__ __align__(16) __bf16 Kt[32 * 64];      // [j][d]  4KB
  __shared__ __align__(16) __bf16 Vs[64 * 32];      // [d][j]  4KB
  __shared__ __align__(16) __bf16 Pl[4][16 * 32];   // per-wave P tile

  const int tid  = threadIdx.x;
  const int lane = tid & 31, wave = tid >> 5;
  const int blk  = blockIdx.x;                      // B*H*(NQ/64) = 512
  const int qblk = blk & 15;
  const int h    = (blk >> 4) & (N_HEADS - 1);
  const int b    = blk >> 7;
  const int half = lane >> 4, col = lane & 15;

  const int qrow0 = b * NQ + qblk * 64 + wave * 16;
  const __bf16* Abase = qb + (size_t)qrow0 * INNER + h * D_HEAD;
  const v16bf qa0 = load_frag(Abase, INNER, lane);       // d = 0..31
  const v16bf qa1 = load_frag(Abase + 32, INNER, lane);  // d = 32..63

  const __bf16* Kbase = kb + (size_t)b * NKV * INNER + h * D_HEAD;
  const __bf16* Vbase = vt + ((size_t)b * INNER + h * D_HEAD) * NKV;
  const unsigned char* mb = mask + (size_t)b * NKV;
  __bf16* P = &Pl[wave][0];

  // staging chunk ids: K tile 32x64 -> 256 x 16B chunks; V tile 64x32 same.
  const int c0 = tid * 2, c1 = tid * 2 + 1;
  const int kr0 = c0 >> 3, kc0 = (c0 & 7) * 8;
  const int kr1 = c1 >> 3, kc1 = (c1 & 7) * 8;
  const int vr0 = c0 >> 2, vc0 = (c0 & 3) * 8;
  const int vr1 = c1 >> 2, vc1 = (c1 & 3) * 8;

  v8f o0, o1, o2, o3;
  float m[8], l[8];
#pragma unroll
  for (int r = 0; r < 8; ++r) {
    o0[r] = o1[r] = o2[r] = o3[r] = 0.0f;
    m[r] = -3.0e38f;
    l[r] = 0.0f;
  }
  const float scale = 0.125f;   // 1/sqrt(64)
  const float NEG = -3.0e38f;

  for (int j0 = 0; j0 < NKV; j0 += 32) {
    __syncthreads();   // previous iteration's LDS reads complete
    // ---- cooperative stage of K[32,64] and V^T[64,32] into LDS ----
    stage_b128(Kbase + (size_t)(j0 + kr0) * INNER + kc0, Kt + kr0 * 64 + kc0);
    stage_b128(Kbase + (size_t)(j0 + kr1) * INNER + kc1, Kt + kr1 * 64 + kc1);
    stage_b128(Vbase + (size_t)vr0 * NKV + j0 + vc0, Vs + vr0 * 32 + vc0);
    stage_b128(Vbase + (size_t)vr1 * NKV + j0 + vc1, Vs + vr1 * 32 + vc1);
    stage_wait();
    __syncthreads();

    // ---- S = Q K^T for two 16-col tiles (f32 accum) ----
    v8f s0, s1;
#pragma unroll
    for (int r = 0; r < 8; ++r) { s0[r] = 0.0f; s1[r] = 0.0f; }
    s0 = wmma_bf16(qa0, load_frag(Kt, 64, lane), s0);
    s0 = wmma_bf16(qa1, load_frag(Kt + 32, 64, lane), s0);
    s1 = wmma_bf16(qa0, load_frag(Kt + 16 * 64, 64, lane), s1);
    s1 = wmma_bf16(qa1, load_frag(Kt + 16 * 64 + 32, 64, lane), s1);

    // ---- scale + mask (lane's column fixed across elements) ----
    const bool v0 = mb[j0 + col] != 0;
    const bool v1 = mb[j0 + 16 + col] != 0;
    float sv0[8], sv1[8], rmax[8];
#pragma unroll
    for (int r = 0; r < 8; ++r) {
      sv0[r] = v0 ? s0[r] * scale : NEG;
      sv1[r] = v1 ? s1[r] * scale : NEG;
      rmax[r] = fmaxf(sv0[r], sv1[r]);
    }
#pragma unroll
    for (int off = 1; off < 16; off <<= 1)
#pragma unroll
      for (int r = 0; r < 8; ++r)
        rmax[r] = fmaxf(rmax[r], __shfl_xor(rmax[r], off, 32));

    // ---- online softmax ----
    float p0[8], p1[8], rsum[8];
#pragma unroll
    for (int r = 0; r < 8; ++r) {
      const float mn = fmaxf(m[r], rmax[r]);
      const float corr = __expf(m[r] - mn);
      m[r] = mn;
      l[r] *= corr;
      o0[r] *= corr; o1[r] *= corr; o2[r] *= corr; o3[r] *= corr;
      p0[r] = __expf(sv0[r] - mn);
      p1[r] = __expf(sv1[r] - mn);
      rsum[r] = p0[r] + p1[r];
    }
#pragma unroll
    for (int off = 1; off < 16; off <<= 1)
#pragma unroll
      for (int r = 0; r < 8; ++r)
        rsum[r] += __shfl_xor(rsum[r], off, 32);
#pragma unroll
    for (int r = 0; r < 8; ++r) l[r] += rsum[r];

    // ---- P -> per-wave LDS tile (cross-lane transpose to A layout) ----
#pragma unroll
    for (int r = 0; r < 8; ++r) {
      const int row = half * 8 + r;
      P[row * 32 + col]      = (__bf16)p0[r];
      P[row * 32 + 16 + col] = (__bf16)p1[r];
    }
    asm volatile("s_wait_dscnt 0" ::: "memory");

    // ---- O += P[16x32] * V[32x64]  (V tile in LDS, row = d, K = j) ----
    const v16bf pa = load_frag(P, 32, lane);
    o0 = wmma_bf16(pa, load_frag(Vs + 0 * 16 * 32, 32, lane), o0);
    o1 = wmma_bf16(pa, load_frag(Vs + 1 * 16 * 32, 32, lane), o1);
    o2 = wmma_bf16(pa, load_frag(Vs + 2 * 16 * 32, 32, lane), o2);
    o3 = wmma_bf16(pa, load_frag(Vs + 3 * 16 * 32, 32, lane), o3);
  }

  // ---- normalize, store O (bf16, [B*NQ, INNER]) ----
  __bf16* Obase = ob + (size_t)qrow0 * INNER + h * D_HEAD;
#pragma unroll
  for (int r = 0; r < 8; ++r) {
    const float inv = 1.0f / l[r];
    const int row = half * 8 + r;
    Obase[(size_t)row * INNER + 0  + col] = (__bf16)(o0[r] * inv);
    Obase[(size_t)row * INNER + 16 + col] = (__bf16)(o1[r] * inv);
    Obase[(size_t)row * INNER + 32 + col] = (__bf16)(o2[r] * inv);
    Obase[(size_t)row * INNER + 48 + col] = (__bf16)(o3[r] * inv);
  }
}

// out = O @ Wo^T + bo : M=B*NQ, N=QDIM, K=INNER, fp32 output.
__global__ __launch_bounds__(128)
void proj_o_kernel(const __bf16* __restrict__ ob, const __bf16* __restrict__ wo,
                   const float* __restrict__ bo, float* __restrict__ out) {
  const int lane = threadIdx.x & 31, wave = threadIdx.x >> 5;
  const int m0 = blockIdx.x * 32;
  const int n0 = (blockIdx.y * 4 + wave) * 32;
  v8f acc[4];
  gemm_tile32(ob + (size_t)m0 * INNER, wo + (size_t)n0 * INNER, INNER, INNER, INNER, lane, acc);
  const int half = lane >> 4, col = lane & 15;
#pragma unroll
  for (int mi = 0; mi < 2; ++mi)
#pragma unroll
    for (int ni = 0; ni < 2; ++ni) {
      const float bias = bo[n0 + ni * 16 + col];
#pragma unroll
      for (int r = 0; r < 8; ++r)
        out[(size_t)(m0 + mi * 16 + half * 8 + r) * QDIM + n0 + ni * 16 + col] =
            acc[mi * 2 + ni][r] + bias;
    }
}

// ---------------------------------------------------------------------------
extern "C" void kernel_launch(void* const* d_in, const int* in_sizes, int n_in,
                              void* d_out, int out_size, void* d_ws, size_t ws_size,
                              hipStream_t stream) {
  const float* x    = (const float*)d_in[0];
  const float* ctx  = (const float*)d_in[1];
  const unsigned char* mask = (const unsigned char*)d_in[2];
  const float* Wq   = (const float*)d_in[3];
  const float* Wkv  = (const float*)d_in[4];
  const float* Wo   = (const float*)d_in[5];
  const float* bo   = (const float*)d_in[6];
  float* out = (float*)d_out;

  char* ws = (char*)d_ws;
  size_t off = 0;
  auto alloc = [&](size_t bytes) -> char* {
    char* p = ws + off;
    off += (bytes + 255) & ~(size_t)255;
    return p;
  };
  __bf16* xb   = (__bf16*)alloc((size_t)BATCH * NQ  * QDIM * 2);
  __bf16* cb   = (__bf16*)alloc((size_t)BATCH * NKV * QDIM * 2);
  __bf16* wqb  = (__bf16*)alloc((size_t)INNER * QDIM * 2);
  __bf16* wkvb = (__bf16*)alloc((size_t)2 * INNER * QDIM * 2);
  __bf16* wob  = (__bf16*)alloc((size_t)QDIM * INNER * 2);
  __bf16* qb   = (__bf16*)alloc((size_t)BATCH * NQ  * INNER * 2);
  __bf16* kb   = (__bf16*)alloc((size_t)BATCH * NKV * INNER * 2);
  __bf16* vt   = (__bf16*)alloc((size_t)BATCH * INNER * NKV * 2);
  __bf16* ob   = (__bf16*)alloc((size_t)BATCH * NQ  * INNER * 2);

  // 1) conversions (2-wide)
  {
    int n2;
    n2 = BATCH * NQ * QDIM / 2;
    cvt_f32_bf16x2<<<(n2 + 255) / 256, 256, 0, stream>>>((const float2*)x, (unsigned int*)xb, n2);
    n2 = BATCH * NKV * QDIM / 2;
    cvt_f32_bf16x2<<<(n2 + 255) / 256, 256, 0, stream>>>((const float2*)ctx, (unsigned int*)cb, n2);
    n2 = INNER * QDIM / 2;
    cvt_f32_bf16x2<<<(n2 + 255) / 256, 256, 0, stream>>>((const float2*)Wq, (unsigned int*)wqb, n2);
    n2 = 2 * INNER * QDIM / 2;
    cvt_f32_bf16x2<<<(n2 + 255) / 256, 256, 0, stream>>>((const float2*)Wkv, (unsigned int*)wkvb, n2);
    n2 = QDIM * INNER / 2;
    cvt_f32_bf16x2<<<(n2 + 255) / 256, 256, 0, stream>>>((const float2*)Wo, (unsigned int*)wob, n2);
  }
  // 2) Q projection: M=4096, N=512 -> grid (128, 4)
  proj_q_kernel<<<dim3(BATCH * NQ / 32, INNER / 128), 128, 0, stream>>>(xb, wqb, qb);
  // 3) KV projection: M=16384, N=1024 -> grid (512, 8)
  proj_kv_kernel<<<dim3(BATCH * NKV / 32, 2 * INNER / 128), 128, 0, stream>>>(cb, wkvb, kb, vt);
  // 4) flash attention: B*H*(NQ/64) = 512 blocks
  attn_kernel<<<dim3(BATCH * N_HEADS * (NQ / 64)), 128, 0, stream>>>(qb, kb, vt, mask, ob);
  // 5) output projection: grid (128, 4)
  proj_o_kernel<<<dim3(BATCH * NQ / 32, QDIM / 128), 128, 0, stream>>>(ob, wob, bo, out);
}
